// GAT_78864189489910
// MI455X (gfx1250) — compile-verified, matrix operand order checked by reading, not run on hardware
//
#include <hip/hip_runtime.h>
#include <hip/hip_bf16.h>
#include <stdint.h>

#define N_NODES  65536
#define N_EDGES  1048576
#define E_TOT    (N_EDGES + N_NODES)   // edges + self loops = 1,114,112
#define N_GRAPHS 256
#define POOL_DIM 256                   // concat of 128+64+32+32

typedef __attribute__((ext_vector_type(16))) _Float16 v16h;
typedef __attribute__((ext_vector_type(8)))  _Float16 v8h;
typedef __attribute__((ext_vector_type(8)))  float    v8f;

// ---------- order-preserving float<->uint map (for atomicMax on floats) ----------
__device__ __forceinline__ unsigned f2ord(float f) {
  unsigned u = __float_as_uint(f);
  return (u & 0x80000000u) ? ~u : (u | 0x80000000u);
}
__device__ __forceinline__ float ord2f(unsigned u) {
  return __uint_as_float((u & 0x80000000u) ? (u & 0x7FFFFFFFu) : ~u);
}
#define ORD_NEG_INF 0x007FFFFFu        // f2ord(-inf)

// ---------- fills ----------
__global__ void fill_f32(float* p, float v, size_t n) {
  size_t i = (size_t)blockIdx.x * blockDim.x + threadIdx.x;
  if (i < n) p[i] = v;
}
__global__ void fill_u32(unsigned* p, unsigned v, size_t n) {
  size_t i = (size_t)blockIdx.x * blockDim.x + threadIdx.x;
  if (i < n) p[i] = v;
}

// ---------- f32 -> f16 conversion (activations) ----------
__global__ void cvt_f16(const float* __restrict__ in, _Float16* __restrict__ out, size_t n) {
  size_t i = (size_t)blockIdx.x * blockDim.x + threadIdx.x;
  if (i < n) out[i] = (_Float16)in[i];
}
// weights: W[K,C] f32 -> WT[C,K] f16 (transpose so B-fragments are contiguous)
__global__ void cvt_f16_T(const float* __restrict__ W, _Float16* __restrict__ WT,
                          int K, int C) {
  size_t i = (size_t)blockIdx.x * blockDim.x + threadIdx.x;
  if (i >= (size_t)K * C) return;
  int k = (int)(i / C), c = (int)(i % C);
  WT[(size_t)c * K + k] = (_Float16)W[i];
}

// ---------- WMMA GEMM: Y[M,C] = A16[M,K] @ BT16[C,K]^T + b  (f16 mul, f32 acc) ----
// One wave32 per 16x16 output tile; K templated so the loop fully unrolls.
// Fragment layout (ISA 7.12.2, 16-bit A 16x32): lane L -> row M = L%16, half = L/16;
// element j -> K = (j&7) + (j>>3)*16 + half*8  ==> two contiguous 8xf16 (16B) runs.
template <int K>
__global__ void wmma_gemm(const _Float16* __restrict__ A16,
                          const _Float16* __restrict__ BT16,
                          const float* __restrict__ bias, float* __restrict__ Y,
                          int M, int C) {
  const int tilesN  = C >> 4;
  const int numTile = (M >> 4) * tilesN;
  const int waveId  = blockIdx.x * (blockDim.x >> 5) + (threadIdx.x >> 5);
  if (waveId >= numTile) return;                    // wave-uniform exit
  const int tm   = (waveId / tilesN) << 4;
  const int tn   = (waveId % tilesN) << 4;
  const int lane = threadIdx.x & 31;
  const int half = lane >> 4;
  const int mr   = lane & 15;

  const _Float16* arow = A16  + (size_t)(tm + mr) * K + half * 8;  // 16B aligned
  const _Float16* brow = BT16 + (size_t)(tn + mr) * K + half * 8;  // 16B aligned

  v8f acc = {};
#pragma unroll
  for (int k0 = 0; k0 < K; k0 += 32) {
    const v8h a0 = *(const v8h*)(arow + k0);        // k0+half*8   .. +7
    const v8h a1 = *(const v8h*)(arow + k0 + 16);   // k0+16+half*8.. +7
    const v8h b0 = *(const v8h*)(brow + k0);
    const v8h b1 = *(const v8h*)(brow + k0 + 16);
    v16h a, b;
#pragma unroll
    for (int j = 0; j < 8; ++j) { a[j] = a0[j]; a[j + 8] = a1[j];
                                  b[j] = b0[j]; b[j + 8] = b1[j]; }
    acc = __builtin_amdgcn_wmma_f32_16x16x32_f16(false, a, false, b, (short)0, acc,
                                                 false, false);
  }
  const int col = tn + mr;
  const float bb = bias ? bias[col] : 0.0f;
#pragma unroll
  for (int r = 0; r < 8; ++r)
    Y[(size_t)(tm + r + 8 * half) * C + col] = acc[r] + bb;   // confirmed C/D layout
}

// ---------- edge helpers ----------
__device__ __forceinline__ void edge_sd(const int* __restrict__ ei, long long e,
                                        int& s, int& d) {
  if (e < (long long)N_EDGES) { s = ei[e]; d = ei[(size_t)N_EDGES + e]; }
  else                        { s = d = (int)(e - N_EDGES); }   // self loop
}

// logit[e] = att . leakyrelu(XL[src] + XR[dst]); segment-max into mOrd[dst].
__global__ void edge_logit(const float* __restrict__ XL, const float* __restrict__ XR,
                           const float* __restrict__ att, const int* __restrict__ ei,
                           float* __restrict__ elog, unsigned* __restrict__ mOrd, int C) {
  const long long e = (long long)blockIdx.x * (blockDim.x >> 5) + (threadIdx.x >> 5);
  const int lane = threadIdx.x & 31;
  if (e >= (long long)E_TOT) return;
  int s, d; edge_sd(ei, e, s, d);
  float part = 0.f;
  for (int c = lane; c < C; c += 32) {
    float z = XL[(size_t)s * C + c] + XR[(size_t)d * C + c];
    z = (z > 0.f) ? z : 0.2f * z;                   // leaky_relu slope 0.2
    part += z * att[c];
  }
#pragma unroll
  for (int off = 16; off > 0; off >>= 1) part += __shfl_down(part, off, 32);
  if (lane == 0) {
    elog[e] = part;
    atomicMax(&mOrd[d], f2ord(part));
  }
}

// e[e] = exp(logit - m[dst]); segment-sum into ssum[dst].
__global__ void edge_exp(const int* __restrict__ ei, float* __restrict__ elog,
                         const unsigned* __restrict__ mOrd, float* __restrict__ ssum,
                         const float* __restrict__ XL, int C) {
  const long long e = (long long)blockIdx.x * blockDim.x + threadIdx.x;
  if (e >= (long long)E_TOT) return;
  int s, d; edge_sd(ei, e, s, d);
  __builtin_prefetch(&XL[(size_t)s * C], 0, 1);     // warm L2 for the scatter pass
  const float ex = __expf(elog[e] - ord2f(mOrd[d]));
  elog[e] = ex;
  atomicAdd(&ssum[d], ex);
}

// out[dst] += alpha * XL[src].
__global__ void edge_scatter(const float* __restrict__ XL, const int* __restrict__ ei,
                             const float* __restrict__ elog, const float* __restrict__ ssum,
                             float* __restrict__ out, int C) {
  const long long e = (long long)blockIdx.x * (blockDim.x >> 5) + (threadIdx.x >> 5);
  const int lane = threadIdx.x & 31;
  if (e >= (long long)E_TOT) return;
  int s, d; edge_sd(ei, e, s, d);
  const float alpha = elog[e] / (ssum[d] + 1e-16f);
  for (int c = lane; c < C; c += 32)
    atomicAdd(&out[(size_t)d * C + c], alpha * XL[(size_t)s * C + c]);
}

// h = gamma*(relu(h + bias) - mean)*rsqrt(var+1e-5) + beta   (bias may be null)
__global__ void bias_relu_bn(float* __restrict__ h, const float* __restrict__ bias,
                             const float* __restrict__ gamma, const float* __restrict__ beta,
                             const float* __restrict__ mean, const float* __restrict__ var,
                             int C, size_t n) {
  const size_t i = (size_t)blockIdx.x * blockDim.x + threadIdx.x;
  if (i >= n) return;
  const int c = (int)(i % C);
  float v = h[i] + (bias ? bias[c] : 0.0f);
  v = fmaxf(v, 0.0f);
  h[i] = gamma[c] * (v - mean[c]) * __frsqrt_rn(var[c] + 1e-5f) + beta[c];
}

// pooled[batch[node]][off + c] += h[node][c]
__global__ void pool_kernel(const float* __restrict__ h, const int* __restrict__ batch,
                            float* __restrict__ pooled, int C, int off, size_t n) {
  const size_t i = (size_t)blockIdx.x * blockDim.x + threadIdx.x;
  if (i >= n) return;
  const size_t node = i / C;
  const int c = (int)(i % C);
  atomicAdd(&pooled[(size_t)batch[node] * POOL_DIM + off + c], h[i]);
}

// sigmoid + log_softmax over 16 classes; one thread per graph row.
__global__ void head_kernel(const float* __restrict__ hl, float* __restrict__ out) {
  const int r = threadIdx.x;                        // 256 threads = 256 graphs
  float v[16], mx = -INFINITY;
#pragma unroll
  for (int c = 0; c < 16; ++c) { v[c] = hl[r * 16 + c]; mx = fmaxf(mx, v[c]); }
  float s = 0.f;
#pragma unroll
  for (int c = 0; c < 16; ++c) s += __expf(v[c] - mx);
  const float lse = mx + __logf(s);
#pragma unroll
  for (int c = 0; c < 16; ++c) {
    out[r * 16 + c]        = 1.0f / (1.0f + __expf(-v[c]));   // sigmoid
    out[4096 + r * 16 + c] = v[c] - lse;                      // log_softmax
  }
}

// =============================== driver =======================================
extern "C" void kernel_launch(void* const* d_in, const int* in_sizes, int n_in,
                              void* d_out, int out_size, void* d_ws, size_t ws_size,
                              hipStream_t stream) {
  (void)in_sizes; (void)n_in; (void)out_size; (void)ws_size;

  // ---- inputs (setup_inputs dict order; params flattened depth-first) ----
  const float* x     = (const float*)d_in[0];
  const int*   ei    = (const int*)d_in[1];          // [2, N_EDGES]
  const int*   batch = (const int*)d_in[2];          // [N_NODES]
  int p = 3;
  const float *Wl1=(const float*)d_in[p++], *bl1=(const float*)d_in[p++],
              *Wr1=(const float*)d_in[p++], *br1=(const float*)d_in[p++],
              *at1=(const float*)d_in[p++], *gb1=(const float*)d_in[p++];
  const float *g1g=(const float*)d_in[p++], *g1b=(const float*)d_in[p++],
              *g1m=(const float*)d_in[p++], *g1v=(const float*)d_in[p++];
  const float *Wl2=(const float*)d_in[p++], *bl2=(const float*)d_in[p++],
              *Wr2=(const float*)d_in[p++], *br2=(const float*)d_in[p++],
              *at2=(const float*)d_in[p++], *gb2=(const float*)d_in[p++];
  const float *g2g=(const float*)d_in[p++], *g2b=(const float*)d_in[p++],
              *g2m=(const float*)d_in[p++], *g2v=(const float*)d_in[p++];
  const float *Wl3=(const float*)d_in[p++], *bl3=(const float*)d_in[p++],
              *Wr3=(const float*)d_in[p++], *br3=(const float*)d_in[p++],
              *at3=(const float*)d_in[p++], *gb3=(const float*)d_in[p++];
  const float *g3g=(const float*)d_in[p++], *g3b=(const float*)d_in[p++],
              *g3m=(const float*)d_in[p++], *g3v=(const float*)d_in[p++];
  const float *L1W=(const float*)d_in[p++], *L1b=(const float*)d_in[p++];
  const float *g5g=(const float*)d_in[p++], *g5b=(const float*)d_in[p++],
              *g5m=(const float*)d_in[p++], *g5v=(const float*)d_in[p++];
  const float *L2W=(const float*)d_in[p++], *L2b=(const float*)d_in[p++];

  // ---- workspace layout (~142 MiB) ----
  char* ws = (char*)d_ws; size_t o = 0;
  float*     h1     = (float*)(ws + o); o += (size_t)N_NODES * 128 * 4;
  float*     h2     = (float*)(ws + o); o += (size_t)N_NODES * 64  * 4;
  float*     h3     = (float*)(ws + o); o += (size_t)N_NODES * 32  * 4;
  float*     XL     = (float*)(ws + o); o += (size_t)N_NODES * 128 * 4;
  float*     XR     = (float*)(ws + o); o += (size_t)N_NODES * 128 * 4;
  float*     elog   = (float*)(ws + o); o += (size_t)E_TOT * 4;
  unsigned*  mOrd   = (unsigned*)(ws + o); o += (size_t)N_NODES * 4;
  float*     ssum   = (float*)(ws + o); o += (size_t)N_NODES * 4;
  float*     pooled = (float*)(ws + o); o += (size_t)N_GRAPHS * POOL_DIM * 4;
  float*     gmid   = (float*)(ws + o); o += (size_t)N_GRAPHS * 128 * 4;
  float*     hl     = (float*)(ws + o); o += (size_t)N_GRAPHS * 16 * 4;
  _Float16*  X16    = (_Float16*)(ws + o); o += (size_t)N_NODES * 128 * 2;
  _Float16*  W16T   = (_Float16*)(ws + o); o += (size_t)256 * 128 * 2;

  auto cdiv = [](long long a, long long b) { return (unsigned)((a + b - 1) / b); };
  auto gemm = [&](const _Float16* A16, const _Float16* BT16, const float* b, float* Y,
                  int M, int K, int C) {
    const unsigned blks = cdiv((long long)(M >> 4) * (C >> 4), 4);
    switch (K) {
      case 64:  wmma_gemm<64>  <<<blks, 128, 0, stream>>>(A16, BT16, b, Y, M, C); break;
      case 128: wmma_gemm<128> <<<blks, 128, 0, stream>>>(A16, BT16, b, Y, M, C); break;
      case 256: wmma_gemm<256> <<<blks, 128, 0, stream>>>(A16, BT16, b, Y, M, C); break;
    }
  };
  auto prep_w = [&](const float* W, int K, int C) {
    cvt_f16_T<<<cdiv((long long)K * C, 256), 256, 0, stream>>>(W, W16T, K, C);
  };

  auto gat_layer = [&](const float* Hin, int K, int C,
                       const float* Wl, const float* bl, const float* Wr, const float* br,
                       const float* att, const float* gbias,
                       const float* bg, const float* bb, const float* bm, const float* bv,
                       float* Hout) {
    cvt_f16<<<cdiv((long long)N_NODES * K, 256), 256, 0, stream>>>(Hin, X16, (size_t)N_NODES * K);
    prep_w(Wl, K, C); gemm(X16, W16T, bl, XL, N_NODES, K, C);
    prep_w(Wr, K, C); gemm(X16, W16T, br, XR, N_NODES, K, C);
    fill_u32<<<cdiv(N_NODES, 256), 256, 0, stream>>>(mOrd, ORD_NEG_INF, N_NODES);
    fill_f32<<<cdiv(N_NODES, 256), 256, 0, stream>>>(ssum, 0.f, N_NODES);
    const size_t nc = (size_t)N_NODES * C;
    fill_f32<<<cdiv((long long)nc, 256), 256, 0, stream>>>(Hout, 0.f, nc);
    edge_logit  <<<cdiv((long long)E_TOT * 32, 256), 256, 0, stream>>>(XL, XR, att, ei, elog, mOrd, C);
    edge_exp    <<<cdiv((long long)E_TOT,      256), 256, 0, stream>>>(ei, elog, mOrd, ssum, XL, C);
    edge_scatter<<<cdiv((long long)E_TOT * 32, 256), 256, 0, stream>>>(XL, ei, elog, ssum, Hout, C);
    bias_relu_bn<<<cdiv((long long)nc, 256), 256, 0, stream>>>(Hout, gbias, bg, bb, bm, bv, C, nc);
  };

  // ---- three GAT layers ----
  gat_layer(x,  128, 128, Wl1, bl1, Wr1, br1, at1, gb1, g1g, g1b, g1m, g1v, h1);
  gat_layer(h1, 128,  64, Wl2, bl2, Wr2, br2, at2, gb2, g2g, g2b, g2m, g2v, h2);
  gat_layer(h2,  64,  32, Wl3, bl3, Wr3, br3, at3, gb3, g3g, g3b, g3m, g3v, h3);

  // ---- global-add pooling: concat [h1 | h2 | h3 | h3] -> [256, 256] ----
  fill_f32<<<cdiv((long long)N_GRAPHS * POOL_DIM, 256), 256, 0, stream>>>(
      pooled, 0.f, (size_t)N_GRAPHS * POOL_DIM);
  pool_kernel<<<cdiv((long long)N_NODES * 128, 256), 256, 0, stream>>>(h1, batch, pooled, 128,   0, (size_t)N_NODES * 128);
  pool_kernel<<<cdiv((long long)N_NODES *  64, 256), 256, 0, stream>>>(h2, batch, pooled,  64, 128, (size_t)N_NODES * 64);
  pool_kernel<<<cdiv((long long)N_NODES *  32, 256), 256, 0, stream>>>(h3, batch, pooled,  32, 192, (size_t)N_NODES * 32);
  pool_kernel<<<cdiv((long long)N_NODES *  32, 256), 256, 0, stream>>>(h3, batch, pooled,  32, 224, (size_t)N_NODES * 32);  // h4 == h3

  // ---- MLP head ----
  cvt_f16<<<cdiv((long long)N_GRAPHS * POOL_DIM, 256), 256, 0, stream>>>(
      pooled, X16, (size_t)N_GRAPHS * POOL_DIM);
  prep_w(L1W, 256, 128);
  gemm(X16, W16T, L1b, gmid, N_GRAPHS, 256, 128);
  bias_relu_bn<<<cdiv((long long)N_GRAPHS * 128, 256), 256, 0, stream>>>(
      gmid, nullptr, g5g, g5b, g5m, g5v, 128, (size_t)N_GRAPHS * 128);
  cvt_f16<<<cdiv((long long)N_GRAPHS * 128, 256), 256, 0, stream>>>(
      gmid, X16, (size_t)N_GRAPHS * 128);
  prep_w(L2W, 128, 16);
  gemm(X16, W16T, L2b, hl, N_GRAPHS, 128, 16);

  head_kernel<<<1, 256, 0, stream>>>(hl, (float*)d_out);
}
